// MoE_50465865728398
// MI455X (gfx1250) — matrix operand order repaired and checked by reading, compiled, not run on hardware
//
#include <hip/hip_runtime.h>
#include <hip/hip_bf16.h>

typedef __attribute__((ext_vector_type(16))) __bf16 v16bf;
typedef __attribute__((ext_vector_type(8)))  __bf16 v8bf;
typedef __attribute__((ext_vector_type(8)))  float  v8f;

#define Bdim 2
#define Ldim 2048
#define Ddim 1024
#define Fdim 4096
#define Edim 8
#define NTOK (Bdim*Ldim)            // 4096 tokens
#define MTILE 64                    // tokens per FFN block
#define TILES_PER_E (NTOK/MTILE)    // 64 (worst case: all tokens on one expert)
#define FCHUNK 256                  // F columns per outer iteration (16 waves x 16)

// ---------------- workspace layout (d_ws) ----------------
// int   cnt[Edim]      | float probSum[Edim]
// int   tokL[Edim*NTOK]| float wgtL[Edim*NTOK]

__global__ void zero_kernel(float* __restrict__ out, int n,
                            int* __restrict__ cnt, float* __restrict__ probSum) {
    int i = blockIdx.x * blockDim.x + threadIdx.x;
    if (i < n) out[i] = 0.0f;
    if (blockIdx.x == 0 && threadIdx.x < Edim) {
        cnt[threadIdx.x] = 0;
        probSum[threadIdx.x] = 0.0f;
    }
}

// One wave (32 lanes) per token: gate scores, softmax stats, top-2 routing.
__global__ void gate_kernel(const float* __restrict__ x,
                            const float* __restrict__ Wg,
                            const float* __restrict__ bg,
                            int* __restrict__ cnt, float* __restrict__ probSum,
                            int* __restrict__ tokL, float* __restrict__ wgtL) {
    int wave = threadIdx.x >> 5;
    int lane = threadIdx.x & 31;
    int t = blockIdx.x * 8 + wave;
    if (t >= NTOK) return;

    const float* xr = x + (size_t)t * Ddim;
    float s[Edim];
#pragma unroll
    for (int e = 0; e < Edim; e++) s[e] = 0.0f;
    for (int d = lane; d < Ddim; d += 32) {
        float xv = xr[d];
        const float* wr = Wg + (size_t)d * Edim;
#pragma unroll
        for (int e = 0; e < Edim; e++) s[e] += xv * wr[e];
    }
#pragma unroll
    for (int off = 16; off > 0; off >>= 1) {
#pragma unroll
        for (int e = 0; e < Edim; e++) s[e] += __shfl_xor(s[e], off, 32);
    }
    if (lane == 0) {
#pragma unroll
        for (int e = 0; e < Edim; e++) s[e] += bg[e];
        float mx = s[0];
#pragma unroll
        for (int e = 1; e < Edim; e++) mx = fmaxf(mx, s[e]);
        float Z = 0.0f, pe[Edim];
#pragma unroll
        for (int e = 0; e < Edim; e++) { pe[e] = __expf(s[e] - mx); Z += pe[e]; }
        float invZ = 1.0f / Z;
#pragma unroll
        for (int e = 0; e < Edim; e++) atomicAdd(&probSum[e], pe[e] * invZ);
        int i1 = 0;
#pragma unroll
        for (int e = 1; e < Edim; e++) if (s[e] > s[i1]) i1 = e;
        int i2 = (i1 == 0) ? 1 : 0;
#pragma unroll
        for (int e = 0; e < Edim; e++) if (e != i1 && s[e] > s[i2]) i2 = e;
        float e2 = __expf(s[i2] - s[i1]);
        float w1 = 1.0f / (1.0f + e2);
        float w2 = e2 * w1;
        int slot1 = atomicAdd(&cnt[i1], 1);
        tokL[i1 * NTOK + slot1] = t;  wgtL[i1 * NTOK + slot1] = w1;
        int slot2 = atomicAdd(&cnt[i2], 1);
        tokL[i2 * NTOK + slot2] = t;  wgtL[i2 * NTOK + slot2] = w2;
    }
}

// Branchless GELU (tanh form): ~8 straight-line VALU, no EXEC divergence.
// tanh(u) = 1 - 2/(exp(2u)+1); rcp(inf)=0 and exp(-inf)=0 give correct tails.
__device__ __forceinline__ float gelu_fast(float v) {
    float u  = v * (0.7978845608028654f + 0.035677408136300125f * v * v);
    float t  = __expf(2.0f * u);
    float th = 1.0f - 2.0f * __builtin_amdgcn_rcpf(t + 1.0f);
    return 0.5f * v * (1.0f + th);
}

// A-fragment swizzle for 16-bit 16x32 A tiles (wave32):
// lane = hi*16 + (m&15); element i: K = i + 8*hi + (i>=8 ? 8 : 0)
__device__ __forceinline__ void a_swizzle(int kr /*0..31*/, int& hi, int& i) {
    if (kr < 16) { hi = kr >> 3; i = kr & 7; }
    else { int k2 = kr - 16; hi = k2 >> 3; i = 8 + (k2 & 7); }
}

// Sparse-dispatch fused expert FFN: 64 tokens of one expert per block.
// 16 waves; wave w = column group ng (16 F-cols in FFN1, 64 D-cols in FFN2).
// Each wave computes ALL 4 row-subtiles so every B fragment feeds 4 WMMAs.
__global__ __launch_bounds__(512, 1)
void ffn_kernel(const float* __restrict__ x,
                const float* __restrict__ W1, const float* __restrict__ b1,
                const float* __restrict__ W2, const float* __restrict__ b2,
                const int* __restrict__ cnt, const int* __restrict__ tokL,
                const float* __restrict__ wgtL, float* __restrict__ out) {
    int e    = blockIdx.x / TILES_PER_E;
    int tile = blockIdx.x % TILES_PER_E;
    int n_e  = cnt[e];
    int row0 = tile * MTILE;
    if (row0 >= n_e) return;                       // uniform early-out

    // x tile bf16, A-fragment order: [msub(4)][kstep(32)][lane(32)][elem(16)]
    __shared__ __bf16 xA[4 * 32 * 32 * 16];        // 128 KB
    // h chunk (64 x 256) A-fragment order: [msub(4)][kstep(8)][lane(32)][elem(16)]
    __shared__ __bf16 hA[4 * 8 * 32 * 16];         // 32 KB  (total = 160 KB exactly)

    const int tid  = threadIdx.x;
    const int ng   = tid >> 5;     // wave id = column group 0..15
    const int lane = tid & 31;
    const int nloc = lane & 15;    // B/C fragment column
    const int khi  = lane >> 4;    // B/C fragment half select
    const int nrem = n_e - row0;   // valid rows in this tile (>=1)

    const int*   tokLe = tokL + e * NTOK + row0;
    const float* wgtLe = wgtL + e * NTOK + row0;

    // Stage 64 x-rows into LDS as swizzled bf16 A fragments.
    // Any aligned 8-run of d maps to 8 contiguous halfwords -> wide ops:
    // 2x global_load_b128 -> 4x v_cvt_pk_bf16_f32 -> 1x ds_store_b128.
    for (int idx = tid; idx < MTILE * (Ddim / 8); idx += 512) {
        int m  = idx >> 7;         // row 0..63   (128 chunks per row)
        int c  = idx & 127;
        int d0 = c << 3;
        int tok = (m < nrem) ? tokLe[m] : 0;
        const float4* xp = (const float4*)(x + (size_t)tok * Ddim + d0);
        float4 f0 = xp[0], f1 = xp[1];
        int msub = m >> 4, mr = m & 15;
        int kk = d0 >> 5, kr0 = d0 & 31;             // kr0 in {0,8,16,24}
        int ln = (((kr0 & 8) ? 1 : 0) << 4) | mr;
        int ib = (kr0 & 16) ? 8 : 0;
        v8bf hv;
        hv[0] = (__bf16)f0.x; hv[1] = (__bf16)f0.y;
        hv[2] = (__bf16)f0.z; hv[3] = (__bf16)f0.w;
        hv[4] = (__bf16)f1.x; hv[5] = (__bf16)f1.y;
        hv[6] = (__bf16)f1.z; hv[7] = (__bf16)f1.w;
        *(v8bf*)&xA[((((msub * 32 + kk) * 32) + ln) << 4) + ib] = hv;
    }
    __syncthreads();

    const float* W1e = W1 + (size_t)e * Ddim * Fdim;
    const float* W2e = W2 + (size_t)e * Fdim * Ddim;

    v8f acc[4][4];                 // [msub][ntile] -> 64 x (ng's 64 D-cols)
#pragma unroll
    for (int m = 0; m < 4; m++)
#pragma unroll
        for (int nt = 0; nt < 4; nt++) acc[m][nt] = (v8f)0.0f;

#pragma unroll 1
    for (int fc = 0; fc < Fdim; fc += FCHUNK) {
        // L2 warm-up for the next F-chunk of both weight streams
        if (fc + FCHUNK < Fdim) {
            __builtin_prefetch(W1e + (size_t)(tid * 2) * Fdim + (fc + FCHUNK), 0, 1);
            __builtin_prefetch(W2e + (size_t)(fc + FCHUNK + (tid >> 1)) * Ddim
                                   + (size_t)(tid & 1) * 512, 0, 1);
        }
        // ---- FFN1: h[64 x 16] (this wave's cols) over K = D; B reused x4 ----
        v8f hacc[4];
#pragma unroll
        for (int m = 0; m < 4; m++) hacc[m] = (v8f)0.0f;
        const int fcol = fc + ng * 16 + nloc;
#pragma unroll 1
        for (int kk = 0; kk < 32; kk++) {
            v16bf a0 = *(const v16bf*)&xA[(((0 * 32 + kk) * 32 + lane) << 4)];
            v16bf a1 = *(const v16bf*)&xA[(((1 * 32 + kk) * 32 + lane) << 4)];
            v16bf a2 = *(const v16bf*)&xA[(((2 * 32 + kk) * 32 + lane) << 4)];
            v16bf a3 = *(const v16bf*)&xA[(((3 * 32 + kk) * 32 + lane) << 4)];
            v16bf bb;
            const int kbase = kk * 32 + khi * 8;
#pragma unroll
            for (int i = 0; i < 16; i++)
                bb[i] = (__bf16)W1e[(size_t)(kbase + i + (i >= 8 ? 8 : 0)) * Fdim + fcol];
            hacc[0] = __builtin_amdgcn_wmma_f32_16x16x32_bf16(false, a0, false, bb, (short)0, hacc[0], false, false);
            hacc[1] = __builtin_amdgcn_wmma_f32_16x16x32_bf16(false, a1, false, bb, (short)0, hacc[1], false, false);
            hacc[2] = __builtin_amdgcn_wmma_f32_16x16x32_bf16(false, a2, false, bb, (short)0, hacc[2], false, false);
            hacc[3] = __builtin_amdgcn_wmma_f32_16x16x32_bf16(false, a3, false, bb, (short)0, hacc[3], false, false);
        }
        // bias + GELU (bias uniform across the 8 C rows)
        const float bias1 = b1[e * Fdim + fcol];
#pragma unroll
        for (int m = 0; m < 4; m++)
#pragma unroll
            for (int r = 0; r < 8; r++) hacc[m][r] = gelu_fast(hacc[m][r] + bias1);

        __syncthreads();   // previous FFN2 reads of hA complete
        // scatter h C-fragments into hA A-fragment layout
        {
            int Kc = ng * 16 + nloc;            // h column -> FFN2 K (0..255)
            int kk2 = Kc >> 5, kr = Kc & 31;
            int hi2, i2;  a_swizzle(kr, hi2, i2);
#pragma unroll
            for (int m = 0; m < 4; m++)
#pragma unroll
                for (int r = 0; r < 8; r++) {
                    int mloc = r + 8 * khi;     // C row within subtile
                    int ln2 = (hi2 << 4) | mloc;
                    hA[(((m * 8 + kk2) * 32 + ln2) << 4) + i2] = (__bf16)hacc[m][r];
                }
        }
        __syncthreads();

        // ---- FFN2: acc[64 x 64 cols] += h[64 x 256] @ W2[fc..fc+256, :] ----
#pragma unroll 1
        for (int kk2 = 0; kk2 < 8; kk2++) {
            v16bf ha0 = *(const v16bf*)&hA[(((0 * 8 + kk2) * 32 + lane) << 4)];
            v16bf ha1 = *(const v16bf*)&hA[(((1 * 8 + kk2) * 32 + lane) << 4)];
            v16bf ha2 = *(const v16bf*)&hA[(((2 * 8 + kk2) * 32 + lane) << 4)];
            v16bf ha3 = *(const v16bf*)&hA[(((3 * 8 + kk2) * 32 + lane) << 4)];
            const int kb2 = fc + kk2 * 32 + khi * 8;
#pragma unroll
            for (int nt = 0; nt < 4; nt++) {
                const int n2 = ng * 64 + nt * 16 + nloc;
                v16bf b2v;
#pragma unroll
                for (int i = 0; i < 16; i++)
                    b2v[i] = (__bf16)W2e[(size_t)(kb2 + i + (i >= 8 ? 8 : 0)) * Ddim + n2];
                acc[0][nt] = __builtin_amdgcn_wmma_f32_16x16x32_bf16(false, ha0, false, b2v, (short)0, acc[0][nt], false, false);
                acc[1][nt] = __builtin_amdgcn_wmma_f32_16x16x32_bf16(false, ha1, false, b2v, (short)0, acc[1][nt], false, false);
                acc[2][nt] = __builtin_amdgcn_wmma_f32_16x16x32_bf16(false, ha2, false, b2v, (short)0, acc[2][nt], false, false);
                acc[3][nt] = __builtin_amdgcn_wmma_f32_16x16x32_bf16(false, ha3, false, b2v, (short)0, acc[3][nt], false, false);
            }
        }
    }

    // ---- weighted combine into out ----
#pragma unroll
    for (int m = 0; m < 4; m++) {
#pragma unroll
        for (int r = 0; r < 8; r++) {
            int row = m * 16 + r + 8 * khi;
            if (row < nrem) {
                float wv  = wgtLe[row];
                int   tok = tokLe[row];
#pragma unroll
                for (int nt = 0; nt < 4; nt++) {
                    int d = ng * 64 + nt * 16 + nloc;
                    atomicAdd(&out[(size_t)tok * Ddim + d],
                              wv * (acc[m][nt][r] + b2[e * Ddim + d]));
                }
            }
        }
    }
}

__global__ void loss_kernel(const int* __restrict__ cnt,
                            const float* __restrict__ probSum,
                            float* __restrict__ out) {
    if (threadIdx.x == 0 && blockIdx.x == 0) {
        const float invN = 1.0f / (float)NTOK;
        float acc = 0.0f;
#pragma unroll
        for (int e = 0; e < Edim; e++)
            acc += ((float)cnt[e] * invN) * (probSum[e] * invN);
        out[(size_t)NTOK * Ddim] = (float)Edim * acc;
    }
}

extern "C" void kernel_launch(void* const* d_in, const int* in_sizes, int n_in,
                              void* d_out, int out_size, void* d_ws, size_t ws_size,
                              hipStream_t stream) {
    const float* x  = (const float*)d_in[0];
    const float* Wg = (const float*)d_in[1];
    const float* bg = (const float*)d_in[2];
    const float* W1 = (const float*)d_in[3];
    const float* b1 = (const float*)d_in[4];
    const float* W2 = (const float*)d_in[5];
    const float* b2 = (const float*)d_in[6];
    float* out = (float*)d_out;

    int*   cnt     = (int*)d_ws;
    float* probSum = (float*)d_ws + Edim;
    int*   tokL    = (int*)d_ws + 2 * Edim;
    float* wgtL    = (float*)d_ws + 2 * Edim + Edim * NTOK;

    zero_kernel<<<(out_size + 255) / 256, 256, 0, stream>>>(out, out_size, cnt, probSum);
    gate_kernel<<<NTOK / 8, 256, 0, stream>>>(x, Wg, bg, cnt, probSum, tokL, wgtL);
    ffn_kernel<<<Edim * TILES_PER_E, 512, 0, stream>>>(x, W1, b1, W2, b2,
                                                       cnt, tokL, wgtL, out);
    loss_kernel<<<1, 32, 0, stream>>>(cnt, probSum, out);
}
